// GATConvQ_52475910423335
// MI455X (gfx1250) — matrix operand order repaired and checked by reading, not compiled, not run on hardware
//
#include <hip/hip_runtime.h>
#include <hip/hip_bf16.h>

typedef __attribute__((ext_vector_type(16))) _Float16 v16h;
typedef __attribute__((ext_vector_type(8)))  _Float16 v8h;
typedef __attribute__((ext_vector_type(8)))  float    v8f;

#define HEADS 8
#define OUTC  16
#define HC    128   /* HEADS*OUTC == IN_C == 128 */
#define NEG_SLOPE 0.2f

// ---- monotonic float<->uint key for exact atomic float max via atomicMax(u32) ----
__device__ __forceinline__ unsigned fkey(float f) {
    unsigned b = __float_as_uint(f);
    return (b & 0x80000000u) ? ~b : (b | 0x80000000u);
}
__device__ __forceinline__ float fdec(unsigned k) {
    return __uint_as_float((k & 0x80000000u) ? (k & 0x7FFFFFFFu) : ~k);
}
__device__ __forceinline__ float lrelu(float x) { return x > 0.f ? x : NEG_SLOPE * x; }

// ---------------- Phase 0: init out=bias, amax=-inf-key, denom=0 ----------------
__global__ void gat_init(float* __restrict__ out, const float* __restrict__ bias,
                         unsigned* __restrict__ amax, float* __restrict__ denom, int N) {
    int t = blockIdx.x * 256 + threadIdx.x;
    if (t < N * HC) out[t] = bias[t & (HC - 1)];
    if (t < N * HEADS) { amax[t] = 0u; denom[t] = 0.f; }
}

// ---------------- Phase 1: xw = x @ W via v_wmma_f32_16x16x32_f16 ----------------
// Block = 256 threads (8 waves), 128 rows per block. W (transposed) and the x tile
// are staged in LDS as f16; each wave owns a 16-row slab and produces 8 16x16 tiles.
__launch_bounds__(256, 1)
__global__ void gat_gemm(const float* __restrict__ x, const float* __restrict__ w,
                         float* __restrict__ xw, int N) {
    __shared__ __align__(16) _Float16 sW[HC * HC];   // sW[n*128+k] = W[k][n] (transposed)
    __shared__ __align__(16) _Float16 sX[128 * HC];  // sX[m*128+k]

    const int tid = threadIdx.x;
    const int base = blockIdx.x * 128;

    for (int i = tid; i < HC * HC; i += 256) {
        int k = i >> 7, n = i & 127;
        sW[n * HC + k] = (_Float16)w[i];
    }
    for (int i = tid; i < 128 * HC; i += 256) {
        int row = base + (i >> 7);
        int k = i & 127;
        sX[i] = (_Float16)((row < N) ? x[(size_t)row * HC + k] : 0.f);
    }
    __syncthreads();

    const int wave = tid >> 5, lane = tid & 31;
    const int half = lane >> 4, l16 = lane & 15;
    const int m0 = wave * 16;

    v8f c[8] = {};

    #pragma unroll
    for (int kb = 0; kb < 4; ++kb) {
        // A fragment (16x32 f16): lanes 0-15 row M, K = {kb*32+0..7, kb*32+16..23};
        // lanes 16-31 same rows, K shifted by +8 (ISA 7.12.2 16-bit A layout).
        const v8h alo = *(const v8h*)&sX[(m0 + l16) * HC + kb * 32 + half * 8];
        const v8h ahi = *(const v8h*)&sX[(m0 + l16) * HC + kb * 32 + 16 + half * 8];
        v16h a;
        #pragma unroll
        for (int i = 0; i < 8; ++i) { a[i] = alo[i]; a[i + 8] = ahi[i]; }

        #pragma unroll
        for (int nb = 0; nb < 8; ++nb) {
            // B fragment (32x16 f16): lane = column N; lanes 0-15 hold K={0..7,16..23},
            // lanes 16-31 hold K={8..15,24..31} (same K order as A).
            const v8h blo = *(const v8h*)&sW[(nb * 16 + l16) * HC + kb * 32 + half * 8];
            const v8h bhi = *(const v8h*)&sW[(nb * 16 + l16) * HC + kb * 32 + 16 + half * 8];
            v16h b;
            #pragma unroll
            for (int i = 0; i < 8; ++i) { b[i] = blo[i]; b[i + 8] = bhi[i]; }
            c[nb] = __builtin_amdgcn_wmma_f32_16x16x32_f16(
                        false, a, false, b, (short)0, c[nb], false, false);
        }
    }

    // C/D layout: VGPR j -> row m0 + j + 8*half, col nb*16 + l16
    #pragma unroll
    for (int nb = 0; nb < 8; ++nb) {
        #pragma unroll
        for (int j = 0; j < 8; ++j) {
            int row = base + m0 + j + 8 * half;
            if (row < N) xw[(size_t)row * HC + nb * 16 + l16] = c[nb][j];
        }
    }
}

// ---------------- Phase 2: per-node attention dots ----------------
__global__ void gat_dots(const float* __restrict__ xw, const float* __restrict__ att,
                         float* __restrict__ a_dst, float* __restrict__ a_src, int N) {
    int t = blockIdx.x * 256 + threadIdx.x;
    if (t >= N * HEADS) return;
    int h = t & 7;
    int n = t >> 3;
    const float4* p = (const float4*)&xw[(size_t)n * HC + h * OUTC];
    float sd = 0.f, ss = 0.f;
    #pragma unroll
    for (int q = 0; q < 4; ++q) {
        float4 v = p[q];
        const float* ad = &att[h * 32 + q * 4];
        const float* as = &att[h * 32 + 16 + q * 4];
        sd += v.x * ad[0] + v.y * ad[1] + v.z * ad[2] + v.w * ad[3];
        ss += v.x * as[0] + v.y * as[1] + v.z * as[2] + v.w * as[3];
    }
    a_dst[t] = sd;
    a_src[t] = ss;
}

// ---------------- Phase 3: segment max over destinations ----------------
__global__ void gat_emax(const long long* __restrict__ src, const long long* __restrict__ dst,
                         const float* __restrict__ a_dst, const float* __restrict__ a_src,
                         unsigned* __restrict__ amax, int E) {
    int t = blockIdx.x * 256 + threadIdx.x;
    if (t >= E * HEADS) return;
    int e = t >> 3, h = t & 7;
    int s = (int)src[e], d = (int)dst[e];
    float alpha = lrelu(a_dst[d * HEADS + h] + a_src[s * HEADS + h]);
    atomicMax(&amax[d * HEADS + h], fkey(alpha));   // -> global_atomic_max_u32
}

// ---------------- Phase 4: segment exp-sum ----------------
__global__ void gat_esum(const long long* __restrict__ src, const long long* __restrict__ dst,
                         const float* __restrict__ a_dst, const float* __restrict__ a_src,
                         const unsigned* __restrict__ amax, float* __restrict__ denom, int E) {
    int t = blockIdx.x * 256 + threadIdx.x;
    if (t >= E * HEADS) return;
    int e = t >> 3, h = t & 7;
    int s = (int)src[e], d = (int)dst[e];
    float alpha = lrelu(a_dst[d * HEADS + h] + a_src[s * HEADS + h]);
    float m = fdec(amax[d * HEADS + h]);
    atomicAdd(&denom[d * HEADS + h], __expf(alpha - m));  // -> global_atomic_add_f32
}

// ---------------- Phase 5: weighted scatter (wave32 per edge) ----------------
// Each lane owns 4 consecutive channels (lane*4 .. lane*4+3), head = lane>>2.
__global__ void gat_scat(const long long* __restrict__ src, const long long* __restrict__ dst,
                         const float* __restrict__ a_dst, const float* __restrict__ a_src,
                         const unsigned* __restrict__ amax, const float* __restrict__ denom,
                         const float* __restrict__ xw, float* __restrict__ out, int E) {
    int t = blockIdx.x * 256 + threadIdx.x;
    int e = t >> 5;
    if (e >= E) return;
    int lane = t & 31;
    int h = lane >> 2;
    int s = (int)src[e], d = (int)dst[e];
    float alpha = lrelu(a_dst[d * HEADS + h] + a_src[s * HEADS + h]);
    float m = fdec(amax[d * HEADS + h]);
    float wgt = __expf(alpha - m) / denom[d * HEADS + h];
    float4 xv = ((const float4*)xw)[(size_t)s * (HC / 4) + lane];
    float* o = &out[(size_t)d * HC + lane * 4];
    atomicAdd(o + 0, xv.x * wgt);
    atomicAdd(o + 1, xv.y * wgt);
    atomicAdd(o + 2, xv.z * wgt);
    atomicAdd(o + 3, xv.w * wgt);
}

extern "C" void kernel_launch(void* const* d_in, const int* in_sizes, int n_in,
                              void* d_out, int out_size, void* d_ws, size_t ws_size,
                              hipStream_t stream) {
    const float*     x    = (const float*)d_in[0];
    const long long* ei   = (const long long*)d_in[1];   // int64 edge_index [2, E]
    const float*     w    = (const float*)d_in[3];
    const float*     att  = (const float*)d_in[4];
    const float*     bias = (const float*)d_in[5];

    const int N = in_sizes[0] / HC;
    const int E = in_sizes[1] / 2;
    const long long* srcp = ei;
    const long long* dstp = ei + E;

    // workspace layout (all float-sized, base is 256B aligned)
    float*    xw    = (float*)d_ws;                    // N*128
    float*    adst  = xw + (size_t)N * HC;             // N*8
    float*    asrc  = adst + (size_t)N * HEADS;        // N*8
    unsigned* amax  = (unsigned*)(asrc + (size_t)N * HEADS); // N*8
    float*    denom = (float*)(amax + (size_t)N * HEADS);    // N*8

    float* out = (float*)d_out;

    gat_init<<<(N * HC + 255) / 256, 256, 0, stream>>>(out, bias, amax, denom, N);
    gat_gemm<<<(N + 127) / 128, 256, 0, stream>>>(x, w, xw, N);
    gat_dots<<<(N * HEADS + 255) / 256, 256, 0, stream>>>(xw, att, adst, asrc, N);
    gat_emax<<<(E * HEADS + 255) / 256, 256, 0, stream>>>(srcp, dstp, adst, asrc, amax, E);
    gat_esum<<<(E * HEADS + 255) / 256, 256, 0, stream>>>(srcp, dstp, adst, asrc, amax, denom, E);
    gat_scat<<<((E * 32) + 255) / 256, 256, 0, stream>>>(srcp, dstp, adst, asrc, amax, denom, xw, out, E);
}
